// DeformConv2d_86105504350808
// MI455X (gfx1250) — matrix operand order repaired and checked by reading, not compiled, not run on hardware
//
#include <hip/hip_runtime.h>

typedef __attribute__((ext_vector_type(16))) __bf16 v16bf;
typedef __attribute__((ext_vector_type(8)))  __bf16 v8bf;
typedef __attribute__((ext_vector_type(8)))  float  v8f;
typedef __attribute__((ext_vector_type(4)))  float  v4f;
// alignment-relaxed float4 for row segments that start at w-1 (4B aligned only);
// gfx1250 DWORD alignment mode accepts dword-aligned b128.
typedef float v4fu __attribute__((ext_vector_type(4), aligned(4)));

namespace {
constexpr int Bn   = 32;            // batch
constexpr int CI   = 256;
constexpr int Hh   = 64, Ww = 64;
constexpr int CO   = 256;
constexpr int KTOT = CI * 9;        // 2304 : im2col K (k = ci*9 + r*3 + s)
constexpr int NTOT = Bn * Hh * Ww;  // 131072 : im2col N (n = b*4096 + h*64 + w)
constexpr int XSZ  = Bn * CI * Hh * Ww;
constexpr int BM = 128, BN = 128, BK = 32;
constexpr int NITER = KTOT / BK;    // 72
}

// out[co][n] = sum_k W[co][k] * Xcol[k][n] + bias[co]
// OIHW weights flatten to row-major [CO][KTOT] in exactly the im2col k order.
// fp32 emulated via bf16 hi/lo split: acc += Ah*Bh + Al*Bh + Ah*Bl (bf16x3).
// Double-buffered LDS, one barrier per K-step; each staging thread's 16 im2col
// columns are one contiguous row segment -> 4 vector loads + pad selects.
__global__ __launch_bounds__(256) void
deform_conv2d_wmma_bf16x3(const float* __restrict__ x,
                          const float* __restrict__ wgt,
                          const float* __restrict__ bias,
                          float* __restrict__ out)
{
  __shared__ __attribute__((aligned(32))) __bf16 As_hi[2][BM * BK];
  __shared__ __attribute__((aligned(32))) __bf16 As_lo[2][BM * BK];
  __shared__ __attribute__((aligned(32))) __bf16 Bs_hi[2][BN * BK];
  __shared__ __attribute__((aligned(32))) __bf16 Bs_lo[2][BN * BK];

  const int t     = threadIdx.x;
  const int lane  = t & 31;
  const int wv    = t >> 5;      // wave 0..7
  const int wm    = wv >> 1;     // 0..3 : 32-row M strip
  const int wn    = wv & 1;      // 0..1 : 64-col N strip
  const int mrow  = lane & 15;   // A row / B,C column within a 16x16 tile
  const int khalf = lane >> 4;

  const int m0 = blockIdx.y * BM;
  const int n0 = blockIdx.x * BN;

  // staging split: 256 threads, 4096 elements per tile -> 16 each
  const int a_row = t >> 1;          // 0..127
  const int a_kc  = (t & 1) * 16;    // 0 / 16
  const int b_kr  = t >> 3;          // 0..31
  const int b_nc  = (t & 7) * 16;    // 0..112

  // B-side thread invariants: 16 consecutive n = one (b,h) row segment
  const int nb0 = n0 + b_nc;         // multiple of 16
  const int bb  = nb0 >> 12;
  const int hh  = (nb0 & 4095) >> 6;
  const int ww0 = nb0 & 63;          // 0,16,32,48
  const bool leftcol  = (ww0 == 0);
  const bool rightcol = (ww0 == 48);
  const float* xbb = x + (size_t)bb * (CI * Hh * Ww);

  v8f acc[2][4];
  #pragma unroll
  for (int i = 0; i < 2; ++i)
    #pragma unroll
    for (int j = 0; j < 4; ++j)
      acc[i][j] = v8f{};

  // ---- phase 1: global loads for K-slice kk into registers ----
  v4f   areg[4];
  float breg[16];
  auto stage_load = [&](int kk) {
    const float* wp = wgt + (m0 + a_row) * KTOT + kk + a_kc;
    #pragma unroll
    for (int i = 0; i < 4; ++i)
      areg[i] = *(const v4f*)(wp + 4 * i);
    if (kk + BK < KTOT)
      __builtin_prefetch(wp + BK, 0, 1);   // global_prefetch_b8: tile after next

    const int k  = kk + b_kr;
    const int ci = k / 9;
    const int t9 = k - ci * 9;
    const int r  = t9 / 3;
    const int s  = t9 - r * 3;
    const int ih = hh + r - 1;
    const bool ih_ok = (unsigned)ih < (unsigned)Hh;
    const int  ihc   = ih_ok ? ih : 0;
    const float* p = xbb + ci * (Hh * Ww) + ihc * Ww + ww0 + (s - 1);

    // vector row-segment loads; clamp first/last pointer against tensor ends
    const float* p0 = p;
    bool lsh = false;
    if (p0 < x) { p0 = x; lsh = true; }             // only b=ci=ih=0,w0=0,s=0
    const float* p3 = p + 12;
    bool rsh = false;
    if (p3 > x + (XSZ - 4)) { p3 = x + (XSZ - 4); rsh = true; }
    const v4fu v0 = *(const v4fu*)p0;
    const v4fu v1 = *(const v4fu*)(p + 4);
    const v4fu v2 = *(const v4fu*)(p + 8);
    const v4fu v3 = *(const v4fu*)p3;

    float e[16];
    e[0]  = lsh ? 0.0f  : v0[0];
    e[1]  = lsh ? v0[0] : v0[1];
    e[2]  = lsh ? v0[1] : v0[2];
    e[3]  = lsh ? v0[2] : v0[3];
    #pragma unroll
    for (int j = 0; j < 4; ++j) { e[4 + j] = v1[j]; e[8 + j] = v2[j]; }
    e[12] = rsh ? v3[1] : v3[0];
    e[13] = rsh ? v3[2] : v3[1];
    e[14] = rsh ? v3[3] : v3[2];
    e[15] = rsh ? 0.0f  : v3[3];

    // padding predicates: vertical (whole segment), horizontal (ends only)
    const bool zl = (s == 0) && leftcol;
    const bool zr = (s == 2) && rightcol;
    #pragma unroll
    for (int j = 0; j < 16; ++j)
      breg[j] = ih_ok ? e[j] : 0.0f;
    if (zl) breg[0]  = 0.0f;
    if (zr) breg[15] = 0.0f;
  };

  // ---- phase 2: fp32 -> bf16 hi/lo split, store into LDS buffer ----
  auto stage_store = [&](int buf) {
    #pragma unroll
    for (int i = 0; i < 4; ++i)
      #pragma unroll
      for (int j = 0; j < 4; ++j) {
        float  f = areg[i][j];
        __bf16 h = (__bf16)f;
        __bf16 l = (__bf16)(f - (float)h);
        As_hi[buf][a_row * BK + a_kc + 4 * i + j] = h;
        As_lo[buf][a_row * BK + a_kc + 4 * i + j] = l;
      }
    #pragma unroll
    for (int j = 0; j < 16; ++j) {
      float  f = breg[j];
      __bf16 h = (__bf16)f;
      __bf16 l = (__bf16)(f - (float)h);
      Bs_hi[buf][(b_nc + j) * BK + b_kr] = h;
      Bs_lo[buf][(b_nc + j) * BK + b_kr] = l;
    }
  };

  // ---- fragments (ISA 16-bit A/B layouts) + 24 WMMAs from buffer buf ----
  auto compute = [&](int buf) {
    v16bf a_hi[2], a_lo[2];
    #pragma unroll
    for (int mt = 0; mt < 2; ++mt) {
      const int mb = (wm * 32 + mt * 16 + mrow) * BK;
      const v8bf h0 = *(const v8bf*)&As_hi[buf][mb + khalf * 8];
      const v8bf h1 = *(const v8bf*)&As_hi[buf][mb + 16 + khalf * 8];
      const v8bf l0 = *(const v8bf*)&As_lo[buf][mb + khalf * 8];
      const v8bf l1 = *(const v8bf*)&As_lo[buf][mb + 16 + khalf * 8];
      #pragma unroll
      for (int e = 0; e < 8; ++e) {
        a_hi[mt][e] = h0[e];  a_hi[mt][e + 8] = h1[e];
        a_lo[mt][e] = l0[e];  a_lo[mt][e + 8] = l1[e];
      }
    }
    v16bf b_hi[4], b_lo[4];
    #pragma unroll
    for (int nt = 0; nt < 4; ++nt) {
      const int nb = (wn * 64 + nt * 16 + mrow) * BK + khalf * 8;
      const v8bf bh0 = *(const v8bf*)&Bs_hi[buf][nb];
      const v8bf bh1 = *(const v8bf*)&Bs_hi[buf][nb + 16];
      const v8bf bl0 = *(const v8bf*)&Bs_lo[buf][nb];
      const v8bf bl1 = *(const v8bf*)&Bs_lo[buf][nb + 16];
      #pragma unroll
      for (int e = 0; e < 8; ++e) {
        b_hi[nt][e] = bh0[e];  b_hi[nt][e + 8] = bh1[e];
        b_lo[nt][e] = bl0[e];  b_lo[nt][e + 8] = bl1[e];
      }
    }
    #pragma unroll
    for (int nt = 0; nt < 4; ++nt)
      #pragma unroll
      for (int mt = 0; mt < 2; ++mt) {
        acc[mt][nt] = __builtin_amdgcn_wmma_f32_16x16x32_bf16(
            false, a_hi[mt], false, b_hi[nt], (short)0, acc[mt][nt], false, false);
        acc[mt][nt] = __builtin_amdgcn_wmma_f32_16x16x32_bf16(
            false, a_lo[mt], false, b_hi[nt], (short)0, acc[mt][nt], false, false);
        acc[mt][nt] = __builtin_amdgcn_wmma_f32_16x16x32_bf16(
            false, a_hi[mt], false, b_lo[nt], (short)0, acc[mt][nt], false, false);
      }
  };

  // ---- software pipeline: double-buffered LDS, one barrier per K-step ----
  stage_load(0);
  stage_store(0);
  __syncthreads();

  int cur = 0;
  for (int it = 0; it < NITER - 1; ++it) {
    stage_load((it + 1) * BK);   // global loads overlap the WMMA chain below
    compute(cur);
    stage_store(cur ^ 1);
    __syncthreads();
    cur ^= 1;
  }
  compute(cur);

  // ---- epilogue: bias add + coalesced fp32 store ----
  // C/D layout: lane = col (0..15), VGPR v -> row = v + (lane>>4)*8
  #pragma unroll
  for (int mt = 0; mt < 2; ++mt) {
    #pragma unroll
    for (int v = 0; v < 8; ++v) {
      const int co = m0 + wm * 32 + mt * 16 + khalf * 8 + v;
      const float bv = bias[co];
      #pragma unroll
      for (int nt = 0; nt < 4; ++nt) {
        const int n  = n0 + wn * 64 + nt * 16 + mrow;
        const int bi = n >> 12;
        const int hw = n & 4095;
        out[((bi * CO + co) << 12) + hw] = acc[mt][nt][v] + bv;
      }
    }
  }
}

extern "C" void kernel_launch(void* const* d_in, const int* in_sizes, int n_in,
                              void* d_out, int out_size, void* d_ws, size_t ws_size,
                              hipStream_t stream) {
  // inputs: 0:x 1:weight 2:bias 3:offset_w 4:offset_b 5:mask_w 6:mask_b
  // offset/mask never reach the reference output (dead code there) -> skip.
  const float* x    = (const float*)d_in[0];
  const float* wgt  = (const float*)d_in[1];
  const float* bias = (const float*)d_in[2];
  float* out = (float*)d_out;
  (void)in_sizes; (void)n_in; (void)out_size; (void)d_ws; (void)ws_size;

  dim3 grid(NTOT / BN, CO / BM);   // (1024, 2)
  deform_conv2d_wmma_bf16x3<<<grid, dim3(256), 0, stream>>>(x, wgt, bias, out);
}
